// SelfAttentionHead_44719199486163
// MI455X (gfx1250) — matrix operand order repaired and checked by reading, not compiled, not run on hardware
//
#include <hip/hip_runtime.h>

typedef __attribute__((ext_vector_type(16))) _Float16 v16h;
typedef __attribute__((ext_vector_type(8)))  _Float16 v8h;
typedef __attribute__((ext_vector_type(8)))  float    v8f;

#define BATCH 16
#define TLEN  2048
#define CEMB  1024
#define HSZ   128
#define MROWS (BATCH*TLEN)

__device__ __forceinline__ v8f wmma_f16(v16h a, v16h b, v8f c) {
  return __builtin_amdgcn_wmma_f32_16x16x32_f16(false, a, false, b, (short)0, c,
                                                false, false);
}

// 64-byte async copy global -> LDS (4 x b128), tracked by ASYNCcnt
__device__ __forceinline__ void async_b128x4(const _Float16* g, _Float16* l) {
#pragma unroll
  for (int i = 0; i < 4; ++i) {
    unsigned int lo = (unsigned int)(unsigned long long)(l + i * 8);
    asm volatile("global_load_async_to_lds_b128 %0, %1, off"
                 :: "v"(lo), "v"((const void*)(g + i * 8)) : "memory");
  }
}

// ---------------------------------------------------------------------------
// Pass 0: W [K=1024][N=128] fp32  ->  Wt [N=128][K=1024] f16  (q scaled 1/32)
// ---------------------------------------------------------------------------
__global__ void wtrans_kernel(const float* __restrict__ Wq,
                              const float* __restrict__ Wk,
                              const float* __restrict__ Wv,
                              _Float16* __restrict__ wt) {
  int idx = blockIdx.x * blockDim.x + threadIdx.x;   // 0 .. 3*C*H-1
  int w   = idx / (CEMB * HSZ);
  int rem = idx - w * (CEMB * HSZ);
  int k = rem / HSZ, n = rem - k * HSZ;
  const float* W = (w == 0) ? Wq : ((w == 1) ? Wk : Wv);
  float scale = (w == 0) ? 0.03125f : 1.0f;          // fold C^-0.5 into Wq
  wt[(size_t)w * CEMB * HSZ + (size_t)n * CEMB + k] = (_Float16)(W[rem] * scale);
}

// ---------------------------------------------------------------------------
// Pass 1: qkv = x @ W   (128x128 tile per block, K chunks of 32 via LDS)
// q,k stored [B,T,H] f16; v stored transposed [B,H,T] f16
// ---------------------------------------------------------------------------
__global__ void __launch_bounds__(256)
qkv_kernel(const float* __restrict__ x, const _Float16* __restrict__ wt_all,
           _Float16* __restrict__ qbuf, _Float16* __restrict__ kbuf,
           _Float16* __restrict__ vtbuf) {
  __shared__ _Float16 Xl[128][40];   // [m][k]  (pad -> 80B row, 16B aligned)
  __shared__ _Float16 Wl[128][40];   // [n][k]

  const int sel  = blockIdx.y;
  const int M0   = blockIdx.x * 128;
  const int tid  = threadIdx.x;
  const int wave = tid >> 5, lane = tid & 31;
  const int lh = lane & 15, hf = lane >> 4;
  const int kbA = hf * 8;            // A-frag K run base
  const int kbB = hf * 16;           // unused; B frag uses kbA too
  const _Float16* wt = wt_all + (size_t)sel * CEMB * HSZ;

  v8f acc[8] = {};

  const int srow = tid >> 1;
  const int sko  = (tid & 1) * 16;

  for (int kk0 = 0; kk0 < CEMB; kk0 += 32) {
    __syncthreads();
    { // stage x chunk: fp32 -> f16
      const float4* p = reinterpret_cast<const float4*>(
          x + (size_t)(M0 + srow) * CEMB + kk0 + sko);
      _Float16* dst = &Xl[srow][sko];
#pragma unroll
      for (int i = 0; i < 4; ++i) {
        float4 v = p[i];
        dst[i * 4 + 0] = (_Float16)v.x; dst[i * 4 + 1] = (_Float16)v.y;
        dst[i * 4 + 2] = (_Float16)v.z; dst[i * 4 + 3] = (_Float16)v.w;
      }
      if (kk0 + 32 < CEMB)   // hint next chunk into cache (global_prefetch_b8)
        __builtin_prefetch(x + (size_t)(M0 + srow) * CEMB + kk0 + 32 + sko, 0, 0);
    }
    { // stage Wt chunk (already f16)
      const v8h* p = reinterpret_cast<const v8h*>(wt + (size_t)srow * CEMB + kk0 + sko);
      *reinterpret_cast<v8h*>(&Wl[srow][sko])     = p[0];
      *reinterpret_cast<v8h*>(&Wl[srow][sko + 8]) = p[1];
    }
    __syncthreads();

    v16h a;
    {
      const _Float16* xr = &Xl[wave * 16 + lh][0];
      reinterpret_cast<v8h*>(&a)[0] = *reinterpret_cast<const v8h*>(xr + kbA);
      reinterpret_cast<v8h*>(&a)[1] = *reinterpret_cast<const v8h*>(xr + 16 + kbA);
    }
#pragma unroll
    for (int nt = 0; nt < 8; ++nt) {
      v16h b;
      const _Float16* wr = &Wl[nt * 16 + lh][0];
      reinterpret_cast<v8h*>(&b)[0] = *reinterpret_cast<const v8h*>(wr + kbA);
      reinterpret_cast<v8h*>(&b)[1] = *reinterpret_cast<const v8h*>(wr + 16 + kbA);
      acc[nt] = wmma_f16(a, b, acc[nt]);
    }
  }

  if (sel < 2) {
    _Float16* dst = (sel == 0) ? qbuf : kbuf;
#pragma unroll
    for (int nt = 0; nt < 8; ++nt) {
      const int col = nt * 16 + lh;
#pragma unroll
      for (int r = 0; r < 8; ++r) {
        int row = M0 + wave * 16 + r + 8 * hf;
        dst[(size_t)row * HSZ + col] = (_Float16)acc[nt][r];
      }
    }
  } else {
#pragma unroll
    for (int nt = 0; nt < 8; ++nt) {
      const int col = nt * 16 + lh;
#pragma unroll
      for (int r = 0; r < 8; ++r) {
        int row = M0 + wave * 16 + r + 8 * hf;
        int b = row >> 11, t = row & (TLEN - 1);
        vtbuf[(size_t)b * HSZ * TLEN + (size_t)col * TLEN + t] = (_Float16)acc[nt][r];
      }
    }
  }
}

// ---------------------------------------------------------------------------
// Pass 2: flash attention. Block = (batch, 128-query tile); 8 waves x 16 rows.
// Double-buffered K/V^T tiles streamed with GLOBAL_LOAD_ASYNC_TO_LDS.
// ---------------------------------------------------------------------------
__global__ void __launch_bounds__(256)
attn_kernel(const _Float16* __restrict__ qbuf, const _Float16* __restrict__ kbuf,
            const _Float16* __restrict__ vtbuf, float* __restrict__ out) {
  __shared__ _Float16 Kl[2][64][136];    // [buf][key][h]
  __shared__ _Float16 Vl[2][128][72];    // [buf][h][key]  (transposed V)
  __shared__ _Float16 Pl[8][16][72];     // per-wave P staging [m][key]

  const int bi  = blockIdx.x >> 4;
  const int qb  = blockIdx.x & 15;
  const int tid = threadIdx.x;
  const int wave = tid >> 5, lane = tid & 31;
  const int lh = lane & 15, hf = lane >> 4;
  const int kbA = hf * 8, kbB = hf * 16;
  const int qrow0 = qb * 128 + wave * 16;

  // per-thread staging coordinates
  const int krow = tid >> 2, kho = (tid & 3) * 32;   // K tile [64][128]
  const int vh   = tid >> 1, vko = (tid & 1) * 32;   // V^T tile [128][64]

  auto issue_tile = [&](int jb, int buf) {
    const int kb0 = jb * 64;
    async_b128x4(kbuf + ((size_t)bi * TLEN + kb0 + krow) * HSZ + kho,
                 &Kl[buf][krow][kho]);
    async_b128x4(vtbuf + (size_t)bi * HSZ * TLEN + (size_t)vh * TLEN + kb0 + vko,
                 &Vl[buf][vh][vko]);
  };

  // Q fragments
  v16h qf[4];
  {
    const _Float16* qr = qbuf + ((size_t)bi * TLEN + qrow0 + lh) * HSZ;
#pragma unroll
    for (int c = 0; c < 4; ++c) {
      reinterpret_cast<v8h*>(&qf[c])[0] = *reinterpret_cast<const v8h*>(qr + c * 32 + kbA);
      reinterpret_cast<v8h*>(&qf[c])[1] = *reinterpret_cast<const v8h*>(qr + c * 32 + 16 + kbA);
    }
  }

  float mrun[8], lrun[8];
#pragma unroll
  for (int r = 0; r < 8; ++r) { mrun[r] = -__builtin_inff(); lrun[r] = 0.0f; }
  v8f o[8] = {};

  const int nkb = (qb + 1) * 2;                       // 64-wide key blocks
  issue_tile(0, 0);                                   // prologue: tile 0 in flight

  for (int jb = 0; jb < nkb; ++jb) {
    const int cur = jb & 1;
    asm volatile("s_wait_asynccnt 0x0" ::: "memory"); // my part of tile jb landed
    __syncthreads();                                  // everyone's part landed;
                                                      // buf cur^1 no longer read
    if (jb + 1 < nkb) issue_tile(jb + 1, cur ^ 1);    // overlap next tile w/ compute

    // S = Q K^T : 16 queries x 64 keys per wave
    v8f s[4] = {};
#pragma unroll
    for (int c = 0; c < 4; ++c) {
#pragma unroll
      for (int t4 = 0; t4 < 4; ++t4) {
        v16h b;
        const _Float16* kr = &Kl[cur][t4 * 16 + lh][c * 32];
        reinterpret_cast<v8h*>(&b)[0] = *reinterpret_cast<const v8h*>(kr + kbA);
        reinterpret_cast<v8h*>(&b)[1] = *reinterpret_cast<const v8h*>(kr + 16 + kbA);
        s[t4] = wmma_f16(qf[c], b, s[t4]);
      }
    }

    const int kb0 = jb * 64;
    if (kb0 + 63 > qrow0) {  // causal mask (wave-uniform trigger, lane-wise select)
#pragma unroll
      for (int t4 = 0; t4 < 4; ++t4) {
        int key = kb0 + t4 * 16 + lh;
#pragma unroll
        for (int r = 0; r < 8; ++r) {
          int qi = qrow0 + r + 8 * hf;
          if (key > qi) s[t4][r] = -__builtin_inff();
        }
      }
    }

    // online softmax; rows live across the 16-lane half (lane = key index)
#pragma unroll
    for (int r = 0; r < 8; ++r) {
      float mx = fmaxf(fmaxf(s[0][r], s[1][r]), fmaxf(s[2][r], s[3][r]));
#pragma unroll
      for (int d = 1; d < 16; d <<= 1) mx = fmaxf(mx, __shfl_xor(mx, d, 32));
      float mn = fmaxf(mrun[r], mx);
      float sc = __expf(mrun[r] - mn);
      mrun[r] = mn;
      float rs = 0.0f;
#pragma unroll
      for (int t4 = 0; t4 < 4; ++t4) {
        float p = __expf(s[t4][r] - mn);
        s[t4][r] = p;
        rs += p;
      }
#pragma unroll
      for (int d = 1; d < 16; d <<= 1) rs += __shfl_xor(rs, d, 32);
      lrun[r] = lrun[r] * sc + rs;
#pragma unroll
      for (int ht = 0; ht < 8; ++ht) o[ht][r] *= sc;
    }

    // P (D-layout) -> per-wave LDS (row-major) for A-fragment reload
#pragma unroll
    for (int t4 = 0; t4 < 4; ++t4)
#pragma unroll
      for (int r = 0; r < 8; ++r)
        Pl[wave][r + 8 * hf][t4 * 16 + lh] = (_Float16)s[t4][r];
    asm volatile("s_wait_dscnt 0x0" ::: "memory");

    // O += P V
#pragma unroll
    for (int c2 = 0; c2 < 2; ++c2) {
      v16h pa;
      const _Float16* pr = &Pl[wave][lh][c2 * 32];
      reinterpret_cast<v8h*>(&pa)[0] = *reinterpret_cast<const v8h*>(pr + kbA);
      reinterpret_cast<v8h*>(&pa)[1] = *reinterpret_cast<const v8h*>(pr + 16 + kbA);
#pragma unroll
      for (int ht = 0; ht < 8; ++ht) {
        v16h b;
        const _Float16* vr = &Vl[cur][ht * 16 + lh][c2 * 32];
        reinterpret_cast<v8h*>(&b)[0] = *reinterpret_cast<const v8h*>(vr + kbA);
        reinterpret_cast<v8h*>(&b)[1] = *reinterpret_cast<const v8h*>(vr + 16 + kbA);
        o[ht] = wmma_f16(pa, b, o[ht]);
      }
    }
  }

#pragma unroll
  for (int ht = 0; ht < 8; ++ht) {
    const int col = ht * 16 + lh;
#pragma unroll
    for (int r = 0; r < 8; ++r) {
      int row = qrow0 + r + 8 * hf;
      out[((size_t)bi * TLEN + row) * HSZ + col] = o[ht][r] / lrun[r];
    }
  }
}

// ---------------------------------------------------------------------------
extern "C" void kernel_launch(void* const* d_in, const int* in_sizes, int n_in,
                              void* d_out, int out_size, void* d_ws, size_t ws_size,
                              hipStream_t stream) {
  (void)in_sizes; (void)n_in; (void)out_size; (void)ws_size;
  const float* x  = (const float*)d_in[0];
  const float* Wq = (const float*)d_in[1];
  const float* Wk = (const float*)d_in[2];
  const float* Wv = (const float*)d_in[3];
  float* out = (float*)d_out;

  _Float16* wt    = (_Float16*)d_ws;                  // 3*H*C      halves
  _Float16* qbuf  = wt   + (size_t)3 * CEMB * HSZ;    // M*H        halves
  _Float16* kbuf  = qbuf + (size_t)MROWS * HSZ;
  _Float16* vtbuf = kbuf + (size_t)MROWS * HSZ;       // ~26 MB total

  wtrans_kernel<<<(3 * CEMB * HSZ) / 256, 256, 0, stream>>>(Wq, Wk, Wv, wt);
  qkv_kernel<<<dim3(MROWS / 128, 3), 256, 0, stream>>>(x, wt, qbuf, kbuf, vtbuf);
  attn_kernel<<<BATCH * (TLEN / 128), 256, 0, stream>>>(qbuf, kbuf, vtbuf, out);
}
